// TriangleMultiplicationOpti_77575699300522
// MI455X (gfx1250) — matrix operand order repaired and checked by reading, not compiled, not run on hardware
//
#include <hip/hip_runtime.h>
#include <hip/hip_bf16.h>

// ---------------------------------------------------------------------------
// Triangle multiplicative update (outgoing), N=512, C_Z=C_MID=128, gfx1250.
// Memory-bound (~78 GFLOP vs ~0.8GB HBM traffic at f16): f16 intermediates,
// WMMA f32_16x16x32_f16 for all matmuls, K-contiguous layouts so A/B
// fragments load straight, and TDM (tensor_load_to_lds) stages the 128x128
// weight tiles into padded LDS rows, double-buffered against WMMA compute.
// ---------------------------------------------------------------------------

#define NRES 512
#define NN   (512 * 512)
#define CCH  128

typedef __attribute__((ext_vector_type(16))) _Float16 v16h;
typedef __attribute__((ext_vector_type(8)))  _Float16 v8h;
typedef __attribute__((ext_vector_type(8)))  float    v8f;
typedef __attribute__((ext_vector_type(4)))  unsigned v4u;
typedef __attribute__((ext_vector_type(8)))  unsigned v8u;

__device__ __forceinline__ float wred(float v) {
#pragma unroll
  for (int s = 16; s > 0; s >>= 1) v += __shfl_xor(v, s, 32);
  return v;
}

__device__ __forceinline__ float sigm(float x) { return 1.0f / (1.0f + __expf(-x)); }

// A fragment: 16x32 (MxK) f16, ISA 7.12.2 layout.
__device__ __forceinline__ v16h frag_a(const _Float16* base, int ld, int lane) {
  const int m = lane & 15, h = lane >> 4;
  const _Float16* p = base + m * ld + h * 8;
  v8h lo = *(const v8h*)p;
  v8h hi = *(const v8h*)(p + 16);
  v16h a;
#pragma unroll
  for (int e = 0; e < 8; ++e) { a[e] = lo[e]; a[e + 8] = hi[e]; }
  return a;
}

// B fragment: 32x16 (KxN) where B[k][n] = W[n][k], W row-major (k contiguous).
__device__ __forceinline__ v16h frag_b(const _Float16* base, int ld, int lane) {
  const int n = lane & 15, h = lane >> 4;
  const _Float16* p = base + n * ld + h * 8;
  v8h lo = *(const v8h*)p;
  v8h hi = *(const v8h*)(p + 16);
  v16h b;
#pragma unroll
  for (int e = 0; e < 8; ++e) { b[e] = lo[e]; b[e + 8] = hi[e]; }
  return b;
}

// 16 rows (this wave's M-tile) x 128 cols against 128x128 weights in LDS
// (row pitch 136 halfs).  1-deep fragment prefetch keeps DS loads in flight.
__device__ __forceinline__ void gemm_tile8(const _Float16* xrow, const _Float16* wm,
                                           int lane, v8f acc[8]) {
  v16h a = frag_a(xrow, 136, lane);
  v16h b = frag_b(wm, 136, lane);
#pragma unroll
  for (int ks = 0; ks < 4; ++ks) {
    v16h an = a;
    if (ks < 3) an = frag_a(xrow + (ks + 1) * 32, 136, lane);
#pragma unroll
    for (int nt = 0; nt < 8; ++nt) {
      v16h bn = b;
      if (nt < 7)
        bn = frag_b(wm + (nt + 1) * (16 * 136) + ks * 32, 136, lane);
      else if (ks < 3)
        bn = frag_b(wm + (ks + 1) * 32, 136, lane);
      acc[nt] = __builtin_amdgcn_wmma_f32_16x16x32_f16(false, a, false, b,
                                                       (short)0, acc[nt], false, false);
      b = bn;
    }
    a = an;
  }
}

// ---------------------------------------------------------------------------
// TDM: stage one 128x128 f16 weight matrix from global into LDS with hardware
// row padding (pad every 64 DWORDs with 4 DWORDs -> 136-half LDS pitch).
// D# per ISA 8.3/8.4: group0 {count=1, lds_addr, global_addr, type=2},
// group1 {data_size=2B, pad_en, pad_interval=64dw, pad_amount=4dw,
//         tensor 128x128, tile 128x128, dim0_stride=128}.
// Issued by one wave; completion published via s_wait_tensorcnt + barrier.
// ---------------------------------------------------------------------------
__device__ __forceinline__ void tdm_load_w(const _Float16* gsrc, unsigned lds_off) {
  unsigned long long ga = (unsigned long long)(uintptr_t)gsrc;
  v4u g0;
  g0[0] = 1u;                                                // count=1
  g0[1] = lds_off;                                           // lds_addr (bytes)
  g0[2] = (unsigned)(ga & 0xFFFFFFFFu);                      // global_addr[31:0]
  g0[3] = (unsigned)((ga >> 32) & 0x01FFFFFFu) | 0x80000000u; // [56:32] | type=2
  v8u g1;
  g1[0] = (1u << 16) | (1u << 20) | (5u << 22) | (3u << 25); // 2B, pad 64dw+4dw
  g1[1] = (128u & 0xFFFFu) << 16;                            // tensor_dim0=128
  g1[2] = (128u & 0xFFFFu) << 16;                            // tensor_dim1=128
  g1[3] = 128u << 16;                                        // tile_dim0=128
  g1[4] = 128u;                                              // tile_dim1=128
  g1[5] = 128u;                                              // dim0_stride=128
  g1[6] = (16384u & 0xFFFFu) << 16;                          // dim1_stride lo
  g1[7] = 16384u >> 16;                                      // dim1_stride hi
  asm volatile("tensor_load_to_lds %0, %1" ::"s"(g0), "s"(g1) : "memory");
}

#define LDSOFF(p) ((unsigned)(uintptr_t)(p))

// ---------------------------------------------------------------------------
// K0: convert the six 128x128 f32 weight matrices to f16 in workspace
// (order: lp, lg, rp, rg, ga, ow) so the TDM can stage raw f16 tiles.
// ---------------------------------------------------------------------------
__global__ __launch_bounds__(256) void tri_k0(const float* __restrict__ lp,
                                              const float* __restrict__ lg,
                                              const float* __restrict__ rp,
                                              const float* __restrict__ rg,
                                              const float* __restrict__ ga,
                                              const float* __restrict__ ow,
                                              _Float16* __restrict__ wh) {
  const float* srcs[6] = {lp, lg, rp, rg, ga, ow};
  const float* s = srcs[blockIdx.x];
  _Float16* d = wh + (size_t)blockIdx.x * (CCH * CCH);
  for (int i = threadIdx.x; i < CCH * CCH; i += 256) d[i] = (_Float16)s[i];
}

// ---------------------------------------------------------------------------
// K1: input LN + 5 projections fused.  Block = 128 pixels, 8 waves.
// Weight tiles arrive via double-buffered TDM while WMMA consumes the other
// buffer.  Outputs: left_t/right_t as [c][pix] f16, gate as [pix][c] f16.
// ---------------------------------------------------------------------------
__global__ __launch_bounds__(256) void tri_k1(
    const float* __restrict__ act,  const float* __restrict__ mask,
    const float* __restrict__ lnw,  const float* __restrict__ lnb,
    const float* __restrict__ lp_b, const float* __restrict__ lg_b,
    const float* __restrict__ rp_b, const float* __restrict__ rg_b,
    const float* __restrict__ ga_b, const _Float16* __restrict__ wh,
    _Float16* __restrict__ left_t, _Float16* __restrict__ right_t,
    _Float16* __restrict__ gate) {
  __shared__ _Float16 xln[128 * 136];
  __shared__ _Float16 wbuf[2][128 * 136];
  __shared__ _Float16 obuf[128 * 136];
  __shared__ float    mrow[128];

  const int tid = threadIdx.x, lane = tid & 31, w = tid >> 5;
  const int p0 = blockIdx.x * 128;

  // Kick off the first weight DMA before doing anything else.
  if (w == 0 && lane == 0) tdm_load_w(wh, LDSOFF(&wbuf[0][0]));

  // ---- Phase A: LayerNorm, one wave per row (float4 per lane) ----
  {
    float4 gwv = *(const float4*)(lnw + lane * 4);
    float4 gbv = *(const float4*)(lnb + lane * 4);
#pragma unroll 1
    for (int it = 0; it < 16; ++it) {
      int r = it * 8 + w;
      float4 x = *(const float4*)(act + (size_t)(p0 + r) * CCH + lane * 4);
      float s  = x.x + x.y + x.z + x.w;
      float ss = x.x * x.x + x.y * x.y + x.z * x.z + x.w * x.w;
      s = wred(s); ss = wred(ss);
      float mu   = s * 0.0078125f;
      float rstd = rsqrtf(ss * 0.0078125f - mu * mu + 1e-5f);
      _Float16* d = xln + r * 136 + lane * 4;
      d[0] = (_Float16)((x.x - mu) * rstd * gwv.x + gbv.x);
      d[1] = (_Float16)((x.y - mu) * rstd * gwv.y + gbv.y);
      d[2] = (_Float16)((x.z - mu) * rstd * gwv.z + gbv.z);
      d[3] = (_Float16)((x.w - mu) * rstd * gwv.w + gbv.w);
      if (lane == 0) mrow[r] = mask[p0 + r];
    }
  }
  __syncthreads();

  const _Float16* xrow = xln + (w * 16) * 136;
  const int m = lane & 15, h = lane >> 4;

  v8f accP[8], accG[8];
  auto zero8 = [&](v8f* a) {
#pragma unroll
    for (int t = 0; t < 8; ++t)
#pragma unroll
      for (int e = 0; e < 8; ++e) a[t][e] = 0.0f;
  };

  // Publish buffer `cur` (TDM done) and start DMA of matrix `nxt` into the
  // other buffer.  nxt < 0 -> nothing left to fetch.
  auto weight_flip = [&](int nxt, int buf) {
    if (w == 0) {
      __builtin_amdgcn_s_wait_tensorcnt(0);
      if (lane == 0 && nxt >= 0)
        tdm_load_w(wh + (size_t)nxt * (CCH * CCH), LDSOFF(&wbuf[buf][0]));
    }
    __syncthreads();
  };

  auto epi_pair = [&](const float* pb, const float* gb2, _Float16* dstT) {
#pragma unroll
    for (int nt = 0; nt < 8; ++nt) {
      int cc = nt * 16 + m;
      float bp = pb[cc], bg = gb2[cc];
#pragma unroll
      for (int v = 0; v < 8; ++v) {
        int row = w * 16 + v + 8 * h;
        float val = mrow[row] * (accP[nt][v] + bp) * sigm(accG[nt][v] + bg);
        obuf[cc * 136 + row] = (_Float16)val;  // [c][row] staging
      }
    }
    __syncthreads();
    for (int idx = tid; idx < CCH * 128; idx += 256) {
      int cc = idx >> 7, col = idx & 127;
      dstT[(size_t)cc * NN + p0 + col] = obuf[cc * 136 + col];  // coalesced
    }
    __syncthreads();
  };

  // stage 0: left_proj (buf0 ready; prefetch lg -> buf1)
  weight_flip(1, 1);
  zero8(accP);
  gemm_tile8(xrow, wbuf[0], lane, accP);
  // stage 1: left_gate (buf1; prefetch rp -> buf0)
  weight_flip(2, 0);
  zero8(accG);
  gemm_tile8(xrow, wbuf[1], lane, accG);
  epi_pair(lp_b, lg_b, left_t);
  // stage 2: right_proj (buf0; prefetch rg -> buf1)
  weight_flip(3, 1);
  zero8(accP);
  gemm_tile8(xrow, wbuf[0], lane, accP);
  // stage 3: right_gate (buf1; prefetch ga -> buf0)
  weight_flip(4, 0);
  zero8(accG);
  gemm_tile8(xrow, wbuf[1], lane, accG);
  epi_pair(rp_b, rg_b, right_t);
  // stage 4: output gating (buf0)
  weight_flip(-1, 0);
  zero8(accP);
  gemm_tile8(xrow, wbuf[0], lane, accP);
#pragma unroll
  for (int nt = 0; nt < 8; ++nt) {
    int cc = nt * 16 + m;
    float bg = ga_b[cc];
#pragma unroll
    for (int v = 0; v < 8; ++v) {
      int row = w * 16 + v + 8 * h;
      gate[(size_t)(p0 + row) * CCH + cc] = (_Float16)sigm(accP[nt][v] + bg);
    }
  }
}

// ---------------------------------------------------------------------------
// K2: Z_c = L_c * R_c^T, 128 channels; grid(32 i-tiles, 128 c), 8 waves,
// wave owns a 16x64 strip; global-fed fragments with next-k prefetch.
// ---------------------------------------------------------------------------
__global__ __launch_bounds__(256) void tri_k2(const _Float16* __restrict__ left_t,
                                              const _Float16* __restrict__ right_t,
                                              _Float16* __restrict__ z_t) {
  const int lane = threadIdx.x & 31, w = threadIdx.x >> 5;
  const int i0 = blockIdx.x * 16;
  const int c  = blockIdx.y;
  const int j0 = w * 64;
  const _Float16* L = left_t  + (size_t)c * NN + (size_t)i0 * NRES;
  const _Float16* R = right_t + (size_t)c * NN + (size_t)j0 * NRES;

  v8f acc[4];
#pragma unroll
  for (int t = 0; t < 4; ++t)
#pragma unroll
    for (int e = 0; e < 8; ++e) acc[t][e] = 0.0f;

  v16h a  = frag_a(L, NRES, lane);
  v16h b0 = frag_b(R, NRES, lane);
#pragma unroll 2
  for (int kb = 0; kb < NRES; kb += 32) {
    v16h b1 = frag_b(R + 1 * (16 * NRES) + kb, NRES, lane);
    v16h b2 = frag_b(R + 2 * (16 * NRES) + kb, NRES, lane);
    v16h b3 = frag_b(R + 3 * (16 * NRES) + kb, NRES, lane);
    v16h an = a, bn0 = b0;
    if (kb + 32 < NRES) {
      an  = frag_a(L + kb + 32, NRES, lane);
      bn0 = frag_b(R + kb + 32, NRES, lane);
    }
    acc[0] = __builtin_amdgcn_wmma_f32_16x16x32_f16(false, a, false, b0, (short)0, acc[0], false, false);
    acc[1] = __builtin_amdgcn_wmma_f32_16x16x32_f16(false, a, false, b1, (short)0, acc[1], false, false);
    acc[2] = __builtin_amdgcn_wmma_f32_16x16x32_f16(false, a, false, b2, (short)0, acc[2], false, false);
    acc[3] = __builtin_amdgcn_wmma_f32_16x16x32_f16(false, a, false, b3, (short)0, acc[3], false, false);
    a = an; b0 = bn0;
  }

  const int m = lane & 15, h = lane >> 4;
  _Float16* Z = z_t + (size_t)c * NN;
#pragma unroll
  for (int jt = 0; jt < 4; ++jt)
#pragma unroll
    for (int v = 0; v < 8; ++v)
      Z[(size_t)(i0 + v + 8 * h) * NRES + j0 + jt * 16 + m] = (_Float16)acc[jt][v];
}

// ---------------------------------------------------------------------------
// K3: center LN over c + out projection + gating.  Block = 128 pixels.
// out_proj weights arrive by TDM, overlapped with z/gate loads and the LN.
// ---------------------------------------------------------------------------
__global__ __launch_bounds__(256) void tri_k3(const _Float16* __restrict__ z_t,
                                              const _Float16* __restrict__ gate,
                                              const float* __restrict__ c_w,
                                              const float* __restrict__ c_b,
                                              const _Float16* __restrict__ ow_h,
                                              const float* __restrict__ o_b,
                                              float* __restrict__ out) {
  __shared__ _Float16 zb[128 * 130];   // [c][pix]
  __shared__ _Float16 zn[128 * 136];   // [pix][c] layernormed
  __shared__ _Float16 wl[128 * 136];   // out_proj_w f16 (TDM, padded rows)
  __shared__ _Float16 gl[128 * 128];   // gate [pix][c]

  const int tid = threadIdx.x, lane = tid & 31, w = tid >> 5;
  const int p0 = blockIdx.x * 128;

  if (w == 0 && lane == 0) tdm_load_w(ow_h, LDSOFF(&wl[0]));

  const _Float16* gsrc = gate + (size_t)p0 * CCH;
  for (int idx = tid; idx < CCH * 128; idx += 256) gl[idx] = gsrc[idx];
  for (int idx = tid; idx < CCH * 128; idx += 256) {
    int cc = idx >> 7, pix = idx & 127;
    zb[cc * 130 + pix] = z_t[(size_t)cc * NN + p0 + pix];
  }
  __syncthreads();

  // center LN across channels: one wave per pixel, 4 channels per lane
#pragma unroll 1
  for (int it = 0; it < 16; ++it) {
    int q = it * 8 + w;
    float x0 = (float)zb[(lane)      * 130 + q];
    float x1 = (float)zb[(lane + 32) * 130 + q];
    float x2 = (float)zb[(lane + 64) * 130 + q];
    float x3 = (float)zb[(lane + 96) * 130 + q];
    float s  = x0 + x1 + x2 + x3;
    float ss = x0 * x0 + x1 * x1 + x2 * x2 + x3 * x3;
    s = wred(s); ss = wred(ss);
    float mu   = s * 0.0078125f;
    float rstd = rsqrtf(ss * 0.0078125f - mu * mu + 1e-5f);
    zn[q * 136 + lane]      = (_Float16)((x0 - mu) * rstd * c_w[lane]      + c_b[lane]);
    zn[q * 136 + lane + 32] = (_Float16)((x1 - mu) * rstd * c_w[lane + 32] + c_b[lane + 32]);
    zn[q * 136 + lane + 64] = (_Float16)((x2 - mu) * rstd * c_w[lane + 64] + c_b[lane + 64]);
    zn[q * 136 + lane + 96] = (_Float16)((x3 - mu) * rstd * c_w[lane + 96] + c_b[lane + 96]);
  }
  if (w == 0) __builtin_amdgcn_s_wait_tensorcnt(0);  // weights landed
  __syncthreads();

  v8f acc[8];
#pragma unroll
  for (int t = 0; t < 8; ++t)
#pragma unroll
    for (int e = 0; e < 8; ++e) acc[t][e] = 0.0f;
  gemm_tile8(zn + (w * 16) * 136, wl, lane, acc);

  const int m = lane & 15, h = lane >> 4;
#pragma unroll
  for (int nt = 0; nt < 8; ++nt) {
    int cc = nt * 16 + m;
    float bo = o_b[cc];
#pragma unroll
    for (int v = 0; v < 8; ++v) {
      int row = w * 16 + v + 8 * h;
      out[(size_t)(p0 + row) * CCH + cc] =
          (acc[nt][v] + bo) * (float)gl[row * 128 + cc];
    }
  }
}

// ---------------------------------------------------------------------------
extern "C" void kernel_launch(void* const* d_in, const int* in_sizes, int n_in,
                              void* d_out, int out_size, void* d_ws, size_t ws_size,
                              hipStream_t stream) {
  const float* act   = (const float*)d_in[0];
  const float* mask  = (const float*)d_in[1];
  const float* lnw   = (const float*)d_in[2];
  const float* lnb   = (const float*)d_in[3];
  const float* lp_w  = (const float*)d_in[4];
  const float* lp_b  = (const float*)d_in[5];
  const float* rp_w  = (const float*)d_in[6];
  const float* rp_b  = (const float*)d_in[7];
  const float* lg_w  = (const float*)d_in[8];
  const float* lg_b  = (const float*)d_in[9];
  const float* rg_w  = (const float*)d_in[10];
  const float* rg_b  = (const float*)d_in[11];
  const float* c_w   = (const float*)d_in[12];
  const float* c_b   = (const float*)d_in[13];
  const float* o_w   = (const float*)d_in[14];
  const float* o_b   = (const float*)d_in[15];
  const float* ga_w  = (const float*)d_in[16];
  const float* ga_b  = (const float*)d_in[17];

  // ws: left_t(64MB f16) | right_t(64MB) | gate(64MB) | z_t(64MB) | wh(192KB)
  char* ws = (char*)d_ws;
  _Float16* left_t  = (_Float16*)(ws);
  _Float16* right_t = (_Float16*)(ws + ((size_t)64 << 20));
  _Float16* gate    = (_Float16*)(ws + ((size_t)128 << 20));
  _Float16* z_t     = (_Float16*)(ws + ((size_t)192 << 20));
  _Float16* wh      = (_Float16*)(ws + ((size_t)256 << 20));

  tri_k0<<<6, 256, 0, stream>>>(lp_w, lg_w, rp_w, rg_w, ga_w, o_w, wh);
  tri_k1<<<NN / 128, 256, 0, stream>>>(act, mask, lnw, lnb, lp_b, lg_b, rp_b,
                                       rg_b, ga_b, wh, left_t, right_t, gate);
  tri_k2<<<dim3(NRES / 16, CCH), 256, 0, stream>>>(left_t, right_t, z_t);
  tri_k3<<<NN / 128, 256, 0, stream>>>(z_t, gate, c_w, c_b, wh + 5 * (CCH * CCH),
                                       o_b, (float*)d_out);
}
